// MOE_7275674600023
// MI455X (gfx1250) — matrix operand order repaired and checked by reading, not compiled, not run on hardware
//
#include <hip/hip_runtime.h>

typedef __attribute__((ext_vector_type(16))) _Float16     v16h;
typedef __attribute__((ext_vector_type(8)))  float        v8f;
typedef __attribute__((ext_vector_type(4)))  unsigned int u32x4;
typedef __attribute__((ext_vector_type(4)))  int          v4i;
typedef __attribute__((ext_vector_type(4)))  float        f32x4;
typedef __attribute__((ext_vector_type(4)))  _Float16     h4;

#define AS1 __attribute__((address_space(1)))
#define AS3 __attribute__((address_space(3)))

union Frag16 { u32x4 u[2]; v16h h; };

#define WMMA_F16(A, B, C) \
  __builtin_amdgcn_wmma_f32_16x16x32_f16(false, (A), false, (B), (short)0, (C), false, false)

#if __has_builtin(__builtin_amdgcn_global_load_async_to_lds_b128)
#define HAS_ASYNC_LDS 1
#else
#define HAS_ASYNC_LDS 0
#endif

__device__ __forceinline__ void async_wait0() {
#if __has_builtin(__builtin_amdgcn_s_wait_asynccnt)
  __builtin_amdgcn_s_wait_asynccnt(0);
#else
  asm volatile("s_wait_asynccnt 0" ::: "memory");
#endif
}

constexpr int Bn  = 4096;   // batch
constexpr int Dn  = 2048;   // model dim
constexpr int Hn  = 8;      // heads
constexpr int Qn  = 64;     // head dim
constexpr int Nn  = 4096;   // num kv
constexpr int Vn  = 128;    // value dim
constexpr int HQn = Hn * Qn;   // 512
constexpr int BHn = Bn * Hn;   // 32768

// ---------------------------------------------------------------------------
// K0a: partial column sums of values: part[32][128]
// ---------------------------------------------------------------------------
__global__ __launch_bounds__(128) void mem_part_vsum(const float* __restrict__ values,
                                                     float* __restrict__ part) {
  const int v = threadIdx.x;          // 0..127
  const int blk = blockIdx.x;         // 0..31 (128 rows each)
  float acc = 0.f;
  const float* p = values + (size_t)blk * 128 * Vn + v;
#pragma unroll 8
  for (int n = 0; n < 128; ++n) acc += p[n * Vn];
  part[blk * Vn + v] = acc;
}

// ---------------------------------------------------------------------------
// K0b: w2[d][h] = sum_v vmean[v] * Wo[d][h*128+v]   (vmean reduced from part)
// ---------------------------------------------------------------------------
__global__ __launch_bounds__(256) void mem_w2(const float* __restrict__ Wo,
                                              const float* __restrict__ part,
                                              float* __restrict__ w2) {
  __shared__ float vm[Vn];
  if (threadIdx.x < Vn) {
    float acc = 0.f;
#pragma unroll
    for (int b = 0; b < 32; ++b) acc += part[b * Vn + threadIdx.x];
    vm[threadIdx.x] = acc * (1.0f / (float)Nn);
  }
  __syncthreads();
  const int d = blockIdx.x * 256 + threadIdx.x;   // 0..2047
  const float* wr = Wo + (size_t)d * (Hn * Vn);
#pragma unroll
  for (int h = 0; h < Hn; ++h) {
    float acc = 0.f;
    for (int v = 0; v < Vn; ++v) acc += vm[v] * wr[h * Vn + v];
    w2[d * Hn + h] = acc;
  }
}

// ---------------------------------------------------------------------------
// K0c: keys fp32 -> f16
// ---------------------------------------------------------------------------
__global__ __launch_bounds__(256) void mem_keys_cvt(const float* __restrict__ keys,
                                                    _Float16* __restrict__ kf) {
  const int i = (blockIdx.x * 256 + threadIdx.x) * 4;   // grid covers Nn*Qn
  f32x4 f = *(const f32x4*)(keys + i);
  h4 hv = { (_Float16)f.x, (_Float16)f.y, (_Float16)f.z, (_Float16)f.w };
  *(h4*)(kf + i) = hv;
}

// ---------------------------------------------------------------------------
// K1: q-proj GEMM  qf(4096x512,f16) = f16(x) @ f16(Wq)^T
//     128x128 block tile, BK=32, 8 waves (4x2), wave tile 32x64 = 2x4 WMMA
//     fp32 tiles converted in flight; next tiles prefetched 2 K-steps ahead
// ---------------------------------------------------------------------------
__global__ __launch_bounds__(256) void mem_qproj(const float* __restrict__ x,
                                                 const float* __restrict__ Wq,
                                                 _Float16* __restrict__ qf) {
  __shared__ _Float16 As[128][40];   // stride 40 halves (80B) -> conflict-free
  __shared__ _Float16 Bs[128][40];
  const int tid  = threadIdx.x;
  const int lane = tid & 31;
  const int w    = tid >> 5;
  const int wm   = w & 3;            // 0..3 -> m offset wm*32
  const int wn   = w >> 2;           // 0..1 -> n offset wn*64
  const int l16  = lane & 15;
  const int hi   = lane >> 4;        // half-wave select
  const int m0 = blockIdx.y * 128;
  const int n0 = blockIdx.x * 128;
  const int lrow = tid >> 1;         // 0..127
  const int lcol = (tid & 1) * 16;   // 0 or 16 (floats)

  const float* xrow = x  + (size_t)(m0 + lrow) * Dn + lcol;
  const float* wrow = Wq + (size_t)(n0 + lrow) * Dn + lcol;

  v8f acc[2][4];
#pragma unroll
  for (int i = 0; i < 2; ++i)
#pragma unroll
    for (int j = 0; j < 4; ++j) acc[i][j] = (v8f){0,0,0,0,0,0,0,0};

  for (int k0 = 0; k0 < Dn; k0 += 32) {
    __syncthreads();
#pragma unroll
    for (int i = 0; i < 4; ++i) {
      f32x4 fa = *(const f32x4*)(xrow + k0 + i * 4);
      f32x4 fb = *(const f32x4*)(wrow + k0 + i * 4);
      h4 ha = { (_Float16)fa.x, (_Float16)fa.y, (_Float16)fa.z, (_Float16)fa.w };
      h4 hb = { (_Float16)fb.x, (_Float16)fb.y, (_Float16)fb.z, (_Float16)fb.w };
      *(h4*)&As[lrow][lcol + i * 4] = ha;
      *(h4*)&Bs[lrow][lcol + i * 4] = hb;
    }
    __syncthreads();

    // prefetch tiles 2 K-steps ahead (hides fp32 fetch behind WMMA phase)
    if (k0 + 64 < Dn) {
      __builtin_prefetch(xrow + k0 + 64, 0, 1);
      __builtin_prefetch(wrow + k0 + 64, 0, 1);
    }

    // preload ALL fragments into distinct registers so the 12 ds_load_b128s
    // issue up front and dscnt waits stagger instead of serializing
    Frag16 a[2], b[4];
#pragma unroll
    for (int i = 0; i < 2; ++i) {
      const _Float16* ap = &As[wm * 32 + i * 16 + l16][hi * 8];
      a[i].u[0] = *(const u32x4*)ap;          // K = {0..7}+8*hi
      a[i].u[1] = *(const u32x4*)(ap + 16);   // K = {16..23}+8*hi
    }
#pragma unroll
    for (int j = 0; j < 4; ++j) {
      const _Float16* bp = &Bs[wn * 64 + j * 16 + l16][hi * 8];
      b[j].u[0] = *(const u32x4*)bp;          // K = {0..7}+8*hi
      b[j].u[1] = *(const u32x4*)(bp + 16);   // K = {16..23}+8*hi
    }
#pragma unroll
    for (int j = 0; j < 4; ++j)
#pragma unroll
      for (int i = 0; i < 2; ++i)
        acc[i][j] = WMMA_F16(a[i].h, b[j].h, acc[i][j]);
  }

#pragma unroll
  for (int i = 0; i < 2; ++i)
#pragma unroll
    for (int j = 0; j < 4; ++j) {
      const int mbase = m0 + wm * 32 + i * 16 + hi * 8;
      const int n     = n0 + wn * 64 + j * 16 + l16;
#pragma unroll
      for (int r = 0; r < 8; ++r)
        qf[(size_t)(mbase + r) * HQn + n] = (_Float16)acc[i][j][r];
    }
}

// ---------------------------------------------------------------------------
// K2: scores = qf @ kf^T (scaled), s[row] = S * (1/S) with S = sum_n exp(score)
//     per wave: one 16-row tile across all 4096 keys
//     keys double-buffered in LDS via async_to_lds: tile kt+1 DMA overlaps
//     the 16 WMMAs + 64 exps of tile kt; wait+barrier only at tile end
// ---------------------------------------------------------------------------
__global__ __launch_bounds__(256) void mem_attn_sum(const _Float16* __restrict__ qf,
                                                    const _Float16* __restrict__ kf,
                                                    float* __restrict__ s) {
  __shared__ _Float16 Ks[2][128][72];  // stride 72 halves (144B) -> conflict-free
  const int tid  = threadIdx.x;
  const int lane = tid & 31;
  const int w    = tid >> 5;
  const int l16  = lane & 15;
  const int hi   = lane >> 4;
  const int rbase = (blockIdx.x * 8 + w) * 16;

  // A fragments for this wave's 16 rows (K=0..31 and K=32..63)
  const _Float16* qrow = qf + (size_t)(rbase + l16) * Qn;
  Frag16 a0, a1;
  a0.u[0] = *(const u32x4*)(qrow + hi * 8);
  a0.u[1] = *(const u32x4*)(qrow + hi * 8 + 16);
  a1.u[0] = *(const u32x4*)(qrow + 32 + hi * 8);
  a1.u[1] = *(const u32x4*)(qrow + 32 + hi * 8 + 16);

  // issue async stage of one 128-key tile into buffer `buf`
  auto stage = [&](int kt, int buf) {
#pragma unroll
    for (int j = 0; j < 4; ++j) {
      const int idx = tid + j * 256;         // 0..1023 uint4 chunks
      const int row = idx >> 3;              // 0..127
      const int col = (idx & 7) * 8;
      const _Float16* src = kf + (size_t)(kt * 128 + row) * Qn + col;
#if HAS_ASYNC_LDS
      __builtin_amdgcn_global_load_async_to_lds_b128(
          (AS1 v4i*)src, (AS3 v4i*)&Ks[buf][row][col],
          /*offset=*/0, /*cpol=*/0);
#else
      *(u32x4*)&Ks[buf][row][col] = *(const u32x4*)src;
#endif
    }
  };

  stage(0, 0);
#if HAS_ASYNC_LDS
  async_wait0();
#endif
  __syncthreads();

  float sum[8] = {0, 0, 0, 0, 0, 0, 0, 0};

  for (int kt = 0; kt < 32; ++kt) {          // 32 tiles of 128 keys
    const int buf = kt & 1;
    if (kt + 1 < 32) stage(kt + 1, buf ^ 1); // DMA next tile behind compute

    for (int ntl = 0; ntl < 8; ++ntl) {      // 16-key WMMA tiles
      const _Float16* krow = &Ks[buf][ntl * 16 + l16][0];
      Frag16 b0, b1;
      b0.u[0] = *(const u32x4*)(krow + hi * 8);
      b0.u[1] = *(const u32x4*)(krow + hi * 8 + 16);
      b1.u[0] = *(const u32x4*)(krow + 32 + hi * 8);
      b1.u[1] = *(const u32x4*)(krow + 32 + hi * 8 + 16);
      v8f c = {0, 0, 0, 0, 0, 0, 0, 0};
      c = WMMA_F16(a0.h, b0.h, c);
      c = WMMA_F16(a1.h, b1.h, c);
#pragma unroll
      for (int r = 0; r < 8; ++r) sum[r] += __expf(c[r] * 0.125f);
    }

    if (kt + 1 < 32) {
#if HAS_ASYNC_LDS
      async_wait0();                          // wave's async copies done
#endif
      __syncthreads();                        // all waves' copies visible
    }
  }

  // reduce across the 16 lanes holding each row (cols 0..15)
#pragma unroll
  for (int mask = 1; mask < 16; mask <<= 1)
#pragma unroll
    for (int r = 0; r < 8; ++r) sum[r] += __shfl_xor(sum[r], mask, 32);

  if (l16 == 0) {
#pragma unroll
    for (int r = 0; r < 8; ++r) {
      const float S = sum[r];
      s[rbase + hi * 8 + r] = S * (1.0f / S);  // == sum_n attn, to ~1e-7
    }
  }
}

// ---------------------------------------------------------------------------
// K3: out[b][d] = x[b][d] + sum_h s[b][h] * w2[d][h]   (memory bound)
// ---------------------------------------------------------------------------
__global__ __launch_bounds__(256) void mem_epilogue(const float* __restrict__ x,
                                                    const float* __restrict__ s,
                                                    const float* __restrict__ w2,
                                                    float* __restrict__ out) {
  const int b = blockIdx.x;
  __shared__ float sh[Hn];
  if (threadIdx.x < Hn) sh[threadIdx.x] = s[b * Hn + threadIdx.x];
  __syncthreads();
  for (int d = threadIdx.x; d < Dn; d += 256) {
    const float* wr = w2 + d * Hn;
    f32x4 w0 = *(const f32x4*)(wr);
    f32x4 w1 = *(const f32x4*)(wr + 4);
    float acc = x[(size_t)b * Dn + d];
    acc += sh[0] * w0.x + sh[1] * w0.y + sh[2] * w0.z + sh[3] * w0.w;
    acc += sh[4] * w1.x + sh[5] * w1.y + sh[6] * w1.z + sh[7] * w1.w;
    out[(size_t)b * Dn + d] = acc;
  }
}

// ---------------------------------------------------------------------------
extern "C" void kernel_launch(void* const* d_in, const int* in_sizes, int n_in,
                              void* d_out, int out_size, void* d_ws, size_t ws_size,
                              hipStream_t stream) {
  (void)in_sizes; (void)n_in; (void)out_size; (void)ws_size;
  const float* x      = (const float*)d_in[0];
  const float* Wq     = (const float*)d_in[1];
  const float* keys   = (const float*)d_in[2];
  const float* values = (const float*)d_in[3];
  const float* Wo     = (const float*)d_in[4];
  float* out = (float*)d_out;

  char* ws = (char*)d_ws;
  _Float16* qf   = (_Float16*)(ws);                    // 32768*64*2  = 4 MiB
  _Float16* kf   = (_Float16*)(ws + 4194304);          // 4096*64*2   = 512 KiB
  float*    sbh  = (float*)(ws + 4718592);             // 32768*4     = 128 KiB
  float*    w2   = (float*)(ws + 4849664);             // 2048*8*4    = 64 KiB
  float*    part = (float*)(ws + 4915200);             // 32*128*4    = 16 KiB

  mem_part_vsum<<<32, 128, 0, stream>>>(values, part);
  mem_w2<<<Dn / 256, 256, 0, stream>>>(Wo, part, w2);
  mem_keys_cvt<<<(Nn * Qn) / (256 * 4), 256, 0, stream>>>(keys, kf);
  mem_qproj<<<dim3(HQn / 128, Bn / 128), 256, 0, stream>>>(x, Wq, qf);
  mem_attn_sum<<<BHn / (16 * 8), 256, 0, stream>>>(qf, kf, sbh);
  mem_epilogue<<<Bn, 256, 0, stream>>>(x, sbh, w2, out);
}
